// PMTransformer_5291399709422
// MI455X (gfx1250) — matrix operand order repaired and checked by reading, not compiled, not run on hardware
//
#include <hip/hip_runtime.h>

typedef _Float16 f16;
typedef __attribute__((ext_vector_type(16))) _Float16 v16h;
typedef __attribute__((ext_vector_type(8)))  _Float16 v8h;
typedef __attribute__((ext_vector_type(8)))  float    v8f;

#define Bb 16
#define Nn 256
#define Dd 128
#define Hh 8
#define HDm 16
#define PD 64
#define M1 128   // pairs per block in the pair-MLP kernel

// Native (1-ulp) reciprocal: single v_rcp_f32, no IEEE divide expansion.
__device__ __forceinline__ float fast_rcp(float x) { return __builtin_amdgcn_rcpf(x); }

// Branchless fast GELU: x * sigmoid(1.702 x). mul + v_exp_f32 + add + v_rcp_f32 + mul.
__device__ __forceinline__ float gelu_f(float x) {
    return x * fast_rcp(1.0f + __expf(-1.702f * x));
}

__device__ __forceinline__ v8f wmma16(v16h a, v16h b, v8f c) {
    return __builtin_amdgcn_wmma_f32_16x16x32_f16(false, a, false, b, (short)0, c, false, false);
}

// A-matrix 16x32 f16 fragment: lane holds row M=lane&15; elems 0-7 = K[off..off+7],
// elems 8-15 = K[off+16..off+23], where caller passes off = kb + (lane>>4)*8.
__device__ __forceinline__ v16h load_a_frag(const f16* rowptr, int off) {
    union { v16h v; v8h h[2]; } u;
    u.h[0] = *(const v8h*)(rowptr + off);
    u.h[1] = *(const v8h*)(rowptr + off + 16);
    return u.v;
}
// B-matrix 32x16 f16 fragment from W^T[n][k] rows: elems 0-7 = K[off..off+7], elems 8-15 = K[off+16..off+23].
__device__ __forceinline__ v16h load_b_frag(const f16* rowptr, int off) {
    union { v16h v; v8h h[2]; } u;
    u.h[0] = *(const v8h*)(rowptr + off);
    u.h[1] = *(const v8h*)(rowptr + off + 16);
    return u.v;
}
// A fragment built directly from an f32 row in global memory (convert to f16).
__device__ __forceinline__ v16h load_a_frag_f32(const float* rowptr, int off) {
    v16h v;
#pragma unroll
    for (int e = 0; e < 8; ++e) v[e] = (f16)rowptr[off + e];
#pragma unroll
    for (int e = 0; e < 8; ++e) v[8 + e] = (f16)rowptr[off + 16 + e];
    return v;
}

// ---------------- K0: per-particle pt / rapidity / phi ----------------
__global__ __launch_bounds__(256) void precompute_kernel(
    const float* __restrict__ x_pf, float* __restrict__ ptg,
    float* __restrict__ rapg, float* __restrict__ phig) {
    int idx = blockIdx.x * 256 + threadIdx.x;      // 0..B*N-1
    int b = idx >> 8, n = idx & 255;
    float px = x_pf[(b * 4 + 0) * Nn + n];
    float py = x_pf[(b * 4 + 1) * Nn + n];
    float pz = x_pf[(b * 4 + 2) * Nn + n];
    float e  = x_pf[(b * 4 + 3) * Nn + n];
    ptg[idx]  = sqrtf(fmaxf(px * px + py * py, 1e-8f));
    rapg[idx] = 0.5f * log1pf(2.0f * pz / fmaxf(e - pz, 1e-20f));
    phig[idx] = atan2f(py, px);
}

// ---------------- K1: fused pairwise features + 4-layer pair MLP ----------------
// grid = B*N*2 blocks; block handles (b, i, half of j): 128 pairs.
__global__ __launch_bounds__(256) void pair_mlp_kernel(
    const float* __restrict__ x_pf,
    const float* __restrict__ ptg, const float* __restrict__ rapg, const float* __restrict__ phig,
    const float* __restrict__ w0, const float* __restrict__ b0,
    const float* __restrict__ w1, const float* __restrict__ b1,
    const float* __restrict__ w2, const float* __restrict__ b2,
    const float* __restrict__ w3, const float* __restrict__ b3,
    float* __restrict__ attn_bias) {
    __shared__ f16  hA[M1 * PD];       // 16 KB (ping)
    __shared__ f16  hB[M1 * PD];       // 16 KB (pong)
    __shared__ f16  w1t[PD * PD];      // 8 KB, transposed [n][k]
    __shared__ f16  w2t[PD * PD];      // 8 KB
    __shared__ float w0s[4 * PD];
    __shared__ float w3s[PD * Hh];
    __shared__ float b0s[PD], b1s[PD], b2s[PD], b3s[Hh];
    __shared__ float fi[7];            // px,py,pz,e,pt,rap,phi of particle i

    const int tid = threadIdx.x;
    const int blk = blockIdx.x;
    const int jhalf = blk & 1;
    const int bi = blk >> 1;
    const int b = bi / Nn;
    const int i = bi % Nn;

    // pull the weight matrices toward this WGP while we do other setup
    __builtin_prefetch(w1 + tid * 16, 0, 0);
    __builtin_prefetch(w2 + tid * 16, 0, 0);

    for (int t = tid; t < PD * PD; t += 256) {
        int n = t >> 6, k = t & 63;
        w1t[n * PD + k] = (f16)w1[k * PD + n];
        w2t[n * PD + k] = (f16)w2[k * PD + n];
    }
    for (int t = tid; t < 4 * PD; t += 256) w0s[t] = w0[t];
    for (int t = tid; t < PD * Hh; t += 256) w3s[t] = w3[t];
    if (tid < PD) { b0s[tid] = b0[tid]; b1s[tid] = b1[tid]; b2s[tid] = b2[tid]; }
    if (tid < Hh) b3s[tid] = b3[tid];
    if (tid < 4) fi[tid] = x_pf[(b * 4 + tid) * Nn + i];
    if (tid == 4) fi[4] = ptg[bi];
    if (tid == 5) fi[5] = rapg[bi];
    if (tid == 6) fi[6] = phig[bi];
    __syncthreads();

    // pairwise features + layer0 (4 -> 64, scalar; two threads share one pair)
    {
        const int p = tid >> 1;
        const int j = jhalf * M1 + p;
        const int bj = b * Nn + j;
        float pxi = fi[0], pyi = fi[1], pzi = fi[2], ei = fi[3];
        float pti = fi[4], rapi = fi[5], phii = fi[6];
        float pxj = x_pf[(b * 4 + 0) * Nn + j];
        float pyj = x_pf[(b * 4 + 1) * Nn + j];
        float pzj = x_pf[(b * 4 + 2) * Nn + j];
        float ej  = x_pf[(b * 4 + 3) * Nn + j];
        float ptj = ptg[bj], rapj = rapg[bj], phij = phig[bj];

        float d_rap = rapi - rapj;
        float d_phi = phii - phij + 3.14159265358979323846f;
        d_phi = fmodf(d_phi, 6.283185307179586f);
        if (d_phi < 0.0f) d_phi += 6.283185307179586f;
        d_phi -= 3.14159265358979323846f;
        float delta = sqrtf(d_rap * d_rap + d_phi * d_phi);
        float f2v = logf(fmaxf(delta, 1e-8f));                           // lndelta
        float ptmin = fminf(pti, ptj);
        float f0v = logf(fmaxf(ptmin * delta, 1e-8f));                   // lnkt
        float f1v = logf(fmaxf(ptmin / fmaxf(pti + ptj, 1e-8f), 1e-8f)); // lnz
        float es = ei + ej, pxs = pxi + pxj, pys = pyi + pyj, pzs = pzi + pzj;
        float f3v = logf(fmaxf(es * es - pxs * pxs - pys * pys - pzs * pzs, 1e-8f)); // lnm2

        const int ns = (tid & 1) * 32;
#pragma unroll 8
        for (int n = ns; n < ns + 32; ++n) {
            float acc = b0s[n] + f0v * w0s[0 * PD + n] + f1v * w0s[1 * PD + n]
                      + f2v * w0s[2 * PD + n] + f3v * w0s[3 * PD + n];
            hA[p * PD + n] = (f16)gelu_f(acc);
        }
    }
    __syncthreads();

    const int wave = tid >> 5;
    const int lane = tid & 31;
    const int halfL = lane >> 4;
    const int lm = lane & 15;

    // 128x64 @ 64x64 GEMM layer with GELU; 8 waves -> one 16-row m-tile each.
    // A fragments are loaded once per wave and reused across all 4 n-tiles.
    auto layer = [&](const f16* hin, f16* hout, const f16* wt, const float* bias) {
        const int mbase = wave * 16;
        const f16* arow = hin + (mbase + lm) * PD;
        v16h a0 = load_a_frag(arow, 0  + halfL * 8);
        v16h a1 = load_a_frag(arow, 32 + halfL * 8);
#pragma unroll
        for (int ntile = 0; ntile < 4; ++ntile) {
            const f16* brow = wt + (ntile * 16 + lm) * PD;
            v16h bf0 = load_b_frag(brow, 0  + halfL * 8);
            v16h bf1 = load_b_frag(brow, 32 + halfL * 8);
            v8f c = {};
            c = wmma16(a0, bf0, c);
            c = wmma16(a1, bf1, c);
            const int col = ntile * 16 + lm;
            const float bb = bias[col];
#pragma unroll
            for (int r = 0; r < 8; ++r) {
                int row = mbase + r + 8 * halfL;
                hout[row * PD + col] = (f16)gelu_f(c[r] + bb);
            }
        }
    };
    layer(hA, hB, w1t, b1s);
    __syncthreads();
    layer(hB, hA, w2t, b2s);
    __syncthreads();

    // layer3: 64 -> 8 heads, scalar; write attn_bias (B,H,N,N)
    {
        const int p = tid >> 1;
        const int hs = (tid & 1) * 4;
        const int j = jhalf * M1 + p;
        float acc[4];
#pragma unroll
        for (int q = 0; q < 4; ++q) acc[q] = b3s[hs + q];
        for (int k = 0; k < PD; ++k) {
            float hv = (float)hA[p * PD + k];
#pragma unroll
            for (int q = 0; q < 4; ++q) acc[q] += hv * w3s[k * Hh + hs + q];
        }
#pragma unroll
        for (int q = 0; q < 4; ++q) {
            int h = hs + q;
            attn_bias[(((long)b * Hh + h) * Nn + i) * Nn + j] = acc[q];
        }
    }
}

// ---------------- K2: M x 128 @ 128 x 128 GEMM + bias (QKV and output proj) ----------------
// grid = M/64 blocks; flags select f16/f32 input and output.
__global__ __launch_bounds__(256) void gemm_kernel(
    const void* __restrict__ in, int in_is_f16,
    const float* __restrict__ w, const float* __restrict__ bias,
    void* __restrict__ out, int out_is_f16) {
    __shared__ f16 As[64 * Dd];    // 16 KB
    __shared__ f16 Wt[Dd * Dd];    // 32 KB, transposed [n][k]
    __shared__ float bs[Dd];
    const int tid = threadIdx.x;
    const long rowbase = (long)blockIdx.x * 64;

    __builtin_prefetch(w + tid * 64, 0, 0);

    for (int t = tid; t < Dd * Dd; t += 256) {
        int n = t >> 7, k = t & 127;
        Wt[n * Dd + k] = (f16)w[k * Dd + n];
    }
    if (tid < Dd) bs[tid] = bias[tid];
    if (in_is_f16) {
        const f16* inp = (const f16*)in;
        for (int t = tid; t < 64 * Dd; t += 256) As[t] = inp[rowbase * Dd + t];
    } else {
        const float* inp = (const float*)in;
        for (int t = tid; t < 64 * Dd; t += 256) As[t] = (f16)inp[rowbase * Dd + t];
    }
    __syncthreads();

    const int wave = tid >> 5, lane = tid & 31, halfL = lane >> 4, lm = lane & 15;
    const int mtile = wave >> 1;
    const f16* arow = As + (mtile * 16 + lm) * Dd;
    v16h af[4];
#pragma unroll
    for (int kc = 0; kc < 4; ++kc) af[kc] = load_a_frag(arow, kc * 32 + halfL * 8);
#pragma unroll
    for (int nt = 0; nt < 4; ++nt) {
        const int ntile = (wave & 1) * 4 + nt;
        const f16* brow = Wt + (ntile * 16 + lm) * Dd;
        v8f c = {};
#pragma unroll
        for (int kc = 0; kc < 4; ++kc) {
            v16h bf = load_b_frag(brow, kc * 32 + halfL * 8);
            c = wmma16(af[kc], bf, c);
        }
        const int col = ntile * 16 + lm;
        const float bb = bs[col];
#pragma unroll
        for (int r = 0; r < 8; ++r) {
            const long row = rowbase + mtile * 16 + r + 8 * halfL;
            float vr = c[r] + bb;
            if (out_is_f16) ((f16*)out)[row * Dd + col] = (f16)vr;
            else            ((float*)out)[row * Dd + col] = vr;
        }
    }
}

// ---------------- K3: scores = Q K^T / 4 + bias (in-place on bias buffer) ----------------
// grid = B*H blocks.
__global__ __launch_bounds__(256) void scores_kernel(
    const f16* __restrict__ q16, const f16* __restrict__ k16,
    float* __restrict__ scores) {
    __shared__ f16 Qs[Nn * 32];    // 16 KB, HD padded 16->32 with zeros
    __shared__ f16 Ks[Nn * 32];    // 16 KB
    const int tid = threadIdx.x;
    const int bh = blockIdx.x, b = bh >> 3, h = bh & 7;

    for (int t = tid; t < Nn * 32; t += 256) {
        int n = t >> 5, d = t & 31;
        f16 qv = (f16)0.0f, kv = (f16)0.0f;
        if (d < HDm) {
            long idx = (((long)b * Nn + n) * Hh + h) * HDm + d;
            qv = q16[idx]; kv = k16[idx];
        }
        Qs[t] = qv; Ks[t] = kv;
    }
    __syncthreads();

    const int wave = tid >> 5, lane = tid & 31, halfL = lane >> 4, lm = lane & 15;
    float* sbase = scores + (long)bh * Nn * Nn;
#pragma unroll
    for (int mt = 0; mt < 2; ++mt) {
        const int mtile = wave * 2 + mt;
        const f16* arow = Qs + (mtile * 16 + lm) * 32;
        v16h a = load_a_frag(arow, halfL * 8);
        for (int ntile = 0; ntile < 16; ++ntile) {
            const f16* brow = Ks + (ntile * 16 + lm) * 32;
            v16h bf = load_b_frag(brow, halfL * 8);
            v8f c = {};
            c = wmma16(a, bf, c);
            const int col = ntile * 16 + lm;
#pragma unroll
            for (int r = 0; r < 8; ++r) {
                const int row = mtile * 16 + r + 8 * halfL;
                float* p = sbase + (long)row * Nn + col;
                *p = 0.25f * c[r] + *p;
            }
        }
    }
}

// ---------------- K4: row softmax, in place; one wave32 per 256-wide row ----------------
__global__ __launch_bounds__(256) void softmax_kernel(float* __restrict__ scores) {
    const int wave = threadIdx.x >> 5, lane = threadIdx.x & 31;
    float* p = scores + ((long)blockIdx.x * 8 + wave) * Nn;
    float4 v0 = ((float4*)p)[lane];
    float4 v1 = ((float4*)p)[32 + lane];
    float mx = fmaxf(fmaxf(fmaxf(v0.x, v0.y), fmaxf(v0.z, v0.w)),
                     fmaxf(fmaxf(v1.x, v1.y), fmaxf(v1.z, v1.w)));
#pragma unroll
    for (int o = 16; o > 0; o >>= 1) mx = fmaxf(mx, __shfl_xor(mx, o, 32));
    v0.x = __expf(v0.x - mx); v0.y = __expf(v0.y - mx);
    v0.z = __expf(v0.z - mx); v0.w = __expf(v0.w - mx);
    v1.x = __expf(v1.x - mx); v1.y = __expf(v1.y - mx);
    v1.z = __expf(v1.z - mx); v1.w = __expf(v1.w - mx);
    float sm = v0.x + v0.y + v0.z + v0.w + v1.x + v1.y + v1.z + v1.w;
#pragma unroll
    for (int o = 16; o > 0; o >>= 1) sm += __shfl_xor(sm, o, 32);
    float inv = fast_rcp(sm);
    v0.x *= inv; v0.y *= inv; v0.z *= inv; v0.w *= inv;
    v1.x *= inv; v1.y *= inv; v1.z *= inv; v1.w *= inv;
    ((float4*)p)[lane] = v0;
    ((float4*)p)[32 + lane] = v1;
}

// ---------------- K5: out = P @ V per (b,h); grid = B*H blocks ----------------
__global__ __launch_bounds__(256) void attnv_kernel(
    const float* __restrict__ scores, const f16* __restrict__ vv,
    f16* __restrict__ attn_out) {
    __shared__ f16 Vt[HDm * Nn];   // 8 KB, V transposed [d][j]
    const int tid = threadIdx.x;
    const int bh = blockIdx.x, b = bh >> 3, h = bh & 7;
    for (int t = tid; t < HDm * Nn; t += 256) {
        int d = t >> 8, j = t & 255;
        Vt[d * Nn + j] = vv[(((long)b * Nn + j) * Hh + h) * HDm + d];
    }
    __syncthreads();

    const int wave = tid >> 5, lane = tid & 31, halfL = lane >> 4, lm = lane & 15;
    const float* sbase = scores + (long)bh * Nn * Nn;
#pragma unroll
    for (int mt = 0; mt < 2; ++mt) {
        const int mtile = wave * 2 + mt;
        const float* arow = sbase + (long)(mtile * 16 + lm) * Nn;
        const f16* brow = Vt + lm * Nn;     // column d = lm
        v8f c = {};
#pragma unroll
        for (int kb = 0; kb < Nn; kb += 32) {
            v16h a  = load_a_frag_f32(arow, kb + halfL * 8);
            v16h bf = load_b_frag(brow, kb + halfL * 8);
            c = wmma16(a, bf, c);
        }
#pragma unroll
        for (int r = 0; r < 8; ++r) {
            const int row = mtile * 16 + r + 8 * halfL;
            attn_out[(((long)b * Nn + row) * Hh + h) * HDm + lm] = (f16)c[r];
        }
    }
}

extern "C" void kernel_launch(void* const* d_in, const int* in_sizes, int n_in,
                              void* d_out, int out_size, void* d_ws, size_t ws_size,
                              hipStream_t stream) {
    (void)in_sizes; (void)n_in; (void)out_size; (void)ws_size;
    const float* x_pf   = (const float*)d_in[0];
    const float* x_feat = (const float*)d_in[1];
    const float* pw0 = (const float*)d_in[2];  const float* pb0 = (const float*)d_in[3];
    const float* pw1 = (const float*)d_in[4];  const float* pb1 = (const float*)d_in[5];
    const float* pw2 = (const float*)d_in[6];  const float* pb2 = (const float*)d_in[7];
    const float* pw3 = (const float*)d_in[8];  const float* pb3 = (const float*)d_in[9];
    const float* wq = (const float*)d_in[10];  const float* wk = (const float*)d_in[11];
    const float* wv = (const float*)d_in[12];  const float* wo = (const float*)d_in[13];
    const float* bq = (const float*)d_in[14];  const float* bk = (const float*)d_in[15];
    const float* bv = (const float*)d_in[16];  const float* bo = (const float*)d_in[17];

    char* ws = (char*)d_ws;
    float* ptg  = (float*)(ws + 0);                      // B*N f32
    float* rapg = (float*)(ws + (1 << 14));
    float* phig = (float*)(ws + (2 << 14));
    float* sc   = (float*)(ws + (1 << 16));              // B*H*N*N f32 = 33.5 MB (bias -> scores -> probs)
    size_t o_q = (1 << 16) + (size_t)Bb * Hh * Nn * Nn * sizeof(float);
    f16* q16  = (f16*)(ws + o_q);                        // B*N*D f16 each
    f16* k16  = q16 + (size_t)Bb * Nn * Dd;
    f16* v16b = k16 + (size_t)Bb * Nn * Dd;
    f16* ao16 = v16b + (size_t)Bb * Nn * Dd;

    precompute_kernel<<<Bb * Nn / 256, 256, 0, stream>>>(x_pf, ptg, rapg, phig);
    pair_mlp_kernel<<<Bb * Nn * 2, 256, 0, stream>>>(
        x_pf, ptg, rapg, phig, pw0, pb0, pw1, pb1, pw2, pb2, pw3, pb3, sc);
    gemm_kernel<<<Bb * Nn / 64, 256, 0, stream>>>(x_feat, 0, wq, bq, q16, 1);
    gemm_kernel<<<Bb * Nn / 64, 256, 0, stream>>>(x_feat, 0, wk, bk, k16, 1);
    gemm_kernel<<<Bb * Nn / 64, 256, 0, stream>>>(x_feat, 0, wv, bv, v16b, 1);
    scores_kernel<<<Bb * Hh, 256, 0, stream>>>(q16, k16, sc);
    softmax_kernel<<<Bb * Hh * Nn / 8, 256, 0, stream>>>(sc);
    attnv_kernel<<<Bb * Hh, 256, 0, stream>>>(sc, v16b, ao16);
    gemm_kernel<<<Bb * Nn / 64, 256, 0, stream>>>(ao16, 1, wo, bo, d_out, 0);
}
